// GeoVAE_84430467105112
// MI455X (gfx1250) — matrix operand order repaired and checked by reading, not compiled, not run on hardware
//
#include <hip/hip_runtime.h>

// ---------- types ----------
typedef __attribute__((ext_vector_type(16))) __bf16   bf16x16;
typedef __attribute__((ext_vector_type(8)))  float    floatx8;
typedef __attribute__((ext_vector_type(4)))  unsigned int u32x4;

union FragU { u32x4 u[2]; bf16x16 v; };

__device__ __forceinline__ unsigned short f2bf(float f) {
    unsigned int u = __float_as_uint(f);
    u += 0x7FFFu + ((u >> 16) & 1u);          // round-to-nearest-even
    return (unsigned short)(u >> 16);
}

// ---------- implicit-GEMM conv / FC / deconv via WMMA bf16 ----------
// All Hout*Wout and Wout in this model are powers of two -> shift/mask decode.
struct ConvParams {
    int Cin, Hin, Win, Cout, Hout, Wout, R, S, stride, pad;
    int mode;          // 0 = normal conv, 1 = deconv (x2 zero-upsampled input, stride-1)
    int relu_in, relu_out, add_res;
    int M, Ktot;
    int hwShift, wShift;   // log2(Hout*Wout), log2(Wout)
};

__global__ __launch_bounds__(256)
void conv_wmma_kernel(const float* __restrict__ in,
                      const unsigned short* __restrict__ wT,   // bf16 [Cout][Ktot]
                      const float* __restrict__ bias,
                      const float* __restrict__ res,
                      float* __restrict__ out,
                      ConvParams p)
{
    constexpr int TM = 128, TN = 64, TK = 64, LDA = 72, LDB = 72;  // 144B LDS rows
    __shared__ __align__(16) unsigned short sA[TM * LDA];
    __shared__ __align__(16) unsigned short sB[TN * LDB];

    const int m0   = blockIdx.x * TM;
    const int n0   = blockIdx.y * TN;
    const int tid  = threadIdx.x;
    const int lane = tid & 31, wid = tid >> 5;
    const int wm   = wid & 3,  wn  = wid >> 2;        // 4x2 wave grid
    const int RS   = p.R * p.S;
    const int HWm  = (1 << p.hwShift) - 1;
    const int Wm   = (1 << p.wShift) - 1;
    const long CinHW = (long)p.Cin * p.Hin * p.Win;

    // per-thread invariants for staging
    const int kkA = tid & 63;      // fixed A-tile column
    const int rA0 = tid >> 6;      // starting A-tile row, step 4

    const floatx8 vzero = {0.f,0.f,0.f,0.f,0.f,0.f,0.f,0.f};
    floatx8 acc[2][2];
#pragma unroll
    for (int a = 0; a < 2; ++a)
#pragma unroll
        for (int b = 0; b < 2; ++b) acc[a][b] = vzero;

    for (int k0 = 0; k0 < p.Ktot; k0 += TK) {
        // ---- B tile: async DMA rows of pre-transposed bf16 weights into LDS ----
        // 64 rows x 128B = 512 x b128 chunks; 2 chunks per thread. ASYNCcnt-tracked,
        // overlaps with the VALU-heavy A staging below.
        {
            int q = tid;
#pragma unroll
            for (int t = 0; t < 2; ++t, q += 256) {
                const int cc = q >> 3, part = q & 7;
                const int c  = (n0 + cc < p.Cout) ? (n0 + cc) : (p.Cout - 1);
                const unsigned short* gp = wT + (long)c * p.Ktot + k0 + part * 8;
                unsigned ldsb = (unsigned)(unsigned long long)&sB[cc * LDB + part * 8];
                asm volatile("global_load_async_to_lds_b128 %0, %1, off"
                             :: "v"(ldsb), "v"(gp) : "memory");
            }
        }
        // ---- A tile: im2col on the fly (k-decode hoisted, shift/mask row decode) ----
        {
            const int k = k0 + kkA;
            const bool kval = (k < p.Ktot);
            int ci = 0, rr = 0, ss = 0;
            if (kval) { ci = k / RS; const int rs = k - ci * RS; rr = rs / p.S; ss = rs - rr * p.S; }
            const long ciOff = (long)ci * p.Hin * p.Win;
            for (int r = rA0; r < TM; r += 4) {
                const int m = m0 + r;
                float v = 0.f;
                if (kval && m < p.M) {
                    const int n  = m >> p.hwShift;
                    const int pq = m & HWm;
                    const int ph = pq >> p.wShift;
                    const int qw = pq & Wm;
                    int hi, wi; bool ok = true;
                    if (p.mode == 0) {
                        hi = ph * p.stride - p.pad + rr;
                        wi = qw * p.stride - p.pad + ss;
                    } else { // deconv: virtual stride-1 conv over 2x zero-upsampled input
                        const int hv = ph - p.pad + rr, wv = qw - p.pad + ss;
                        ok = ((hv & 1) == 0) && ((wv & 1) == 0);
                        hi = hv >> 1; wi = wv >> 1;
                    }
                    if (ok && hi >= 0 && hi < p.Hin && wi >= 0 && wi < p.Win) {
                        v = in[(long)n * CinHW + ciOff + (long)hi * p.Win + wi];
                        if (p.relu_in) v = fmaxf(v, 0.f);
                    }
                }
                sA[r * LDA + kkA] = f2bf(v);
            }
        }
        asm volatile("s_wait_asynccnt 0x0" ::: "memory");
        __syncthreads();

        // ---- two K=32 WMMA slices per staging round ----
        const int kb = (lane >> 4) << 3;   // A: lanes16-31 offset K by 8

#pragma unroll
        for (int ksl = 0; ksl < 2; ++ksl) {
            const int ko = ksl * 32;
            FragU afr[2], bfr[2];
#pragma unroll
            for (int sm = 0; sm < 2; ++sm) {
                const int row = wm * 32 + sm * 16 + (lane & 15);
                afr[sm].u[0] = *(const u32x4*)&sA[row * LDA + ko + kb];
                afr[sm].u[1] = *(const u32x4*)&sA[row * LDA + ko + kb + 16];
            }
#pragma unroll
            for (int sn = 0; sn < 2; ++sn) {
                const int col = wn * 32 + sn * 16 + (lane & 15);
                bfr[sn].u[0] = *(const u32x4*)&sB[col * LDB + ko + kb];
                bfr[sn].u[1] = *(const u32x4*)&sB[col * LDB + ko + kb + 16];
            }
#pragma unroll
            for (int sm = 0; sm < 2; ++sm)
#pragma unroll
                for (int sn = 0; sn < 2; ++sn)
                    acc[sm][sn] = __builtin_amdgcn_wmma_f32_16x16x32_bf16(
                        false, afr[sm].v, false, bfr[sn].v,
                        (short)0, acc[sm][sn], false, false);
        }
        __syncthreads();
    }

    // ---- epilogue: C layout (VGPR v -> M = v + 8*(lane>=16), N = lane&15) ----
    const int mlBase = (lane >> 4) << 3;
#pragma unroll
    for (int sm = 0; sm < 2; ++sm) {
#pragma unroll
        for (int sn = 0; sn < 2; ++sn) {
            const int c = n0 + wn * 32 + sn * 16 + (lane & 15);
#pragma unroll
            for (int vv = 0; vv < 8; ++vv) {
                const int m = m0 + wm * 32 + sm * 16 + mlBase + vv;
                if (m < p.M && c < p.Cout) {
                    float xv = acc[sm][sn][vv] + bias[c];
                    const int n = m >> p.hwShift, pq = m & HWm;
                    const long oidx = ((long)n * p.Cout + c) * (HWm + 1) + pq;
                    if (p.add_res)  xv += res[oidx];
                    if (p.relu_out) xv = fmaxf(xv, 0.f);
                    out[oidx] = xv;
                }
            }
        }
    }
}

// ---------- weight prep: fp32 -> bf16, stored N-major [Cout][Ktot] ----------
__global__ void wprep_copy(const float* __restrict__ w, unsigned short* __restrict__ o, long n) {
    long i = (long)blockIdx.x * 256 + threadIdx.x;
    if (i < n) o[i] = f2bf(w[i]);          // OIHW flat == [Cout][K]
}
__global__ void wprep_transpose(const float* __restrict__ w, unsigned short* __restrict__ o,
                                int K, int C) {  // src [K][C] -> dst [C][K]
    long i = (long)blockIdx.x * 256 + threadIdx.x;
    const long tot = (long)K * C;
    if (i < tot) {
        int c = (int)(i / K), k = (int)(i - (long)c * K);
        o[i] = f2bf(w[(long)k * C + c]);
    }
}
__global__ void wprep_hwio(const float* __restrict__ w, unsigned short* __restrict__ o,
                           int R, int S, int I, int O) {   // HWIO -> dst [O][I*R*S]
    const int K = I * R * S;
    const long tot = (long)K * O;
    long i = (long)blockIdx.x * 256 + threadIdx.x;
    if (i < tot) {
        int oo = (int)(i / K), k = (int)(i - (long)oo * K);
        int ci = k / (R * S), rs = k - ci * (R * S);
        int rr = rs / S, ss = rs - rr * S;
        o[i] = f2bf(w[(((long)rr * S + ss) * I + ci) * O + oo]);
    }
}

// ---------- small fp32 helpers ----------
__global__ void sgemm_naive(const float* __restrict__ A, const float* __restrict__ B,
                            const float* __restrict__ bias, float* __restrict__ C,
                            int M, int N, int K, int relu) {
    int i = blockIdx.x * blockDim.x + threadIdx.x;
    if (i >= M * N) return;
    int r = i / N, c = i - r * N;
    float s = 0.f;
    for (int k = 0; k < K; ++k) s += A[r * K + k] * B[k * N + c];
    if (bias) s += bias[c];
    if (relu) s = fmaxf(s, 0.f);
    C[i] = s;
}

__global__ void deg_kernel(const int* __restrict__ dst, int E,
                           float* __restrict__ deg, float* __restrict__ dinv, int Nn) {
    int j = threadIdx.x;
    if (j >= Nn) return;
    float d = 1.f;
    for (int e = 0; e < E; ++e) if (dst[e] == j) d += 1.f;
    deg[j] = d; dinv[j] = rsqrtf(d);
}

__global__ void gcn_agg_kernel(const float* __restrict__ xl, const int* __restrict__ src,
                               const int* __restrict__ dst, int E,
                               const float* __restrict__ deg, const float* __restrict__ dinv,
                               const float* __restrict__ bias, float* __restrict__ out,
                               int Nn, int F) {
    int i = blockIdx.x * blockDim.x + threadIdx.x;
    if (i >= Nn * F) return;
    int j = i / F, f = i - j * F;
    float s = 0.f;
    for (int e = 0; e < E; ++e)
        if (dst[e] == j) s += xl[src[e] * F + f] * dinv[src[e]];
    s *= dinv[j];
    s += xl[j * F + f] / deg[j] + bias[f];
    out[i] = fmaxf(s, 0.f);
}

// single-block: thresholded symmetric adjacency (with complete-graph fallback) -> norm_adj
__global__ __launch_bounds__(256)
void dim_adj_kernel(const float* __restrict__ logits, float* __restrict__ adj, int D) {
    __shared__ int ssum;
    __shared__ float sdinv[256];
    const int tid = threadIdx.x;
    int cnt = 0;
    for (int i = tid; i < D * D; i += 256) {
        int r = i / D, c = i - r * D;
        if (r != c && (logits[r * D + c] + logits[c * D + r]) > 0.f) cnt++;
    }
    if (tid == 0) ssum = 0;
    __syncthreads();
    atomicAdd(&ssum, cnt);
    __syncthreads();
    const bool fb = (ssum == 0);
    if (tid < D) {
        float rs = 1.f;
        for (int c = 0; c < D; ++c) {
            if (c == tid) continue;
            float l = logits[tid * D + c] + logits[c * D + tid];
            rs += fb ? 1.f : (l > 0.f ? 1.f : 0.f);
        }
        sdinv[tid] = rsqrtf(rs);
    }
    __syncthreads();
    for (int i = tid; i < D * D; i += 256) {
        int r = i / D, c = i - r * D;
        float a;
        if (r == c) a = 1.f;
        else {
            float l = logits[r * D + c] + logits[c * D + r];
            a = fb ? 1.f : (l > 0.f ? 1.f : 0.f);
        }
        adj[i] = a * sdinv[r] * sdinv[c];
    }
}

__global__ void reparam_kernel(const float* __restrict__ mu, const float* __restrict__ lv,
                               const float* __restrict__ eps, const float* __restrict__ vinst,
                               const float* __restrict__ dimp, float* __restrict__ z,
                               float* __restrict__ out_vinst, int n, int D) {
    int i = blockIdx.x * blockDim.x + threadIdx.x;
    if (i >= n) return;
    float sg = __expf(0.5f * lv[i]);
    float vv = vinst[i] * dimp[i % D];
    z[i] = mu[i] + sg * (eps[i] * vv);
    out_vinst[i] = vinst[i];
}

__global__ void copyf_kernel(const float* __restrict__ s, float* __restrict__ d, int n) {
    int i = blockIdx.x * blockDim.x + threadIdx.x;
    if (i < n) d[i] = s[i];
}

__global__ __launch_bounds__(256)
void elbo_kernel(const float* __restrict__ recon, const float* __restrict__ x, int nrec,
                 const float* __restrict__ z, const float* __restrict__ lv, int nz,
                 float* __restrict__ out_elbo) {
    __shared__ float red[256];
    const int tid = threadIdx.x;
    float s1 = 0.f;
    for (int i = tid; i < nrec; i += 256) { float d = recon[i] - x[i]; s1 += d * d; }
    float s2 = 0.f, s3 = 0.f;
    for (int i = tid; i < nz; i += 256) {
        float zz = z[i]; s2 += zz * zz;
        float sig2 = __expf(lv[i]); s3 += 1.f + logf(sig2 + 1e-8f);
    }
    red[tid] = -s1 - 0.5f * s2 + 0.5f * s3;
    __syncthreads();
    for (int o = 128; o > 0; o >>= 1) { if (tid < o) red[tid] += red[tid + o]; __syncthreads(); }
    if (tid == 0) out_elbo[0] = red[0] - 0.5f * (float)nz * 1.8378770664093453f; // log(2*pi)
}

__global__ __launch_bounds__(256)
void active_kernel(const float* __restrict__ mu, const float* __restrict__ lv,
                   int B, int D, float* __restrict__ out) {
    __shared__ int cnt;
    const int d = threadIdx.x;
    if (threadIdx.x == 0) cnt = 0;
    __syncthreads();
    if (d < D) {
        float s = 0.f;
        for (int b = 0; b < B; ++b) {
            float l = logf(__expf(lv[b * D + d]) + 1e-8f);
            float m = mu[b * D + d];
            s += -0.5f * (1.f + l - m * m - __expf(l));
        }
        if (s / (float)B > 0.01f) atomicAdd(&cnt, 1);
    }
    __syncthreads();
    if (threadIdx.x == 0) out[0] = (float)cnt;
}

// ---------- host ----------
static int ilog2(int v) { int s = 0; while ((1 << s) < v) ++s; return s; }

static void run_conv(hipStream_t s, const float* in, const unsigned short* wT,
                     const float* bias, const float* res, float* out,
                     int Cin, int Hin, int Win, int Cout, int Hout, int Wout,
                     int R, int S, int stride, int pad, int mode, int ri, int ro, int ar,
                     int N) {
    ConvParams p;
    p.Cin = Cin; p.Hin = Hin; p.Win = Win; p.Cout = Cout;
    p.Hout = Hout; p.Wout = Wout; p.R = R; p.S = S; p.stride = stride; p.pad = pad;
    p.mode = mode; p.relu_in = ri; p.relu_out = ro; p.add_res = ar;
    p.M = N * Hout * Wout; p.Ktot = Cin * R * S;
    p.hwShift = ilog2(Hout * Wout); p.wShift = ilog2(Wout);
    dim3 grid((p.M + 127) / 128, (Cout + 63) / 64);
    conv_wmma_kernel<<<grid, dim3(256), 0, s>>>(in, wT, bias, res ? res : out, out, p);
}

extern "C" void kernel_launch(void* const* d_in, const int* in_sizes, int n_in,
                              void* d_out, int out_size, void* d_ws, size_t ws_size,
                              hipStream_t stream) {
    const int B = 256, C = 3, H = 64, D = 256, hid = 128, Rb = 32, L = 2;
    const int f4 = H / 4;                 // 16
    const int FLAT = hid * f4 * f4;       // 32768
    const int E = in_sizes[2] / 2;        // 510

    const float* x        = (const float*)d_in[0];
    const float* eps      = (const float*)d_in[1];
    const int*   ei       = (const int*)d_in[2];
    const int*   esrc = ei, *edst = ei + E;
    const float* ec1w = (const float*)d_in[3];  const float* ec1b = (const float*)d_in[4];
    const float* ec2w = (const float*)d_in[5];  const float* ec2b = (const float*)d_in[6];
    const float* ec3w = (const float*)d_in[7];  const float* ec3b = (const float*)d_in[8];
    const float* er1w = (const float*)d_in[9];  const float* er1b = (const float*)d_in[10];
    const float* er2w = (const float*)d_in[11]; const float* er2b = (const float*)d_in[12];
    const float* fmw  = (const float*)d_in[13]; const float* fmb  = (const float*)d_in[14];
    const float* flw  = (const float*)d_in[15]; const float* flb  = (const float*)d_in[16];
    const float* dfw  = (const float*)d_in[17]; const float* dfb  = (const float*)d_in[18];
    const float* dr1w = (const float*)d_in[19]; const float* dr1b = (const float*)d_in[20];
    const float* dr2w = (const float*)d_in[21]; const float* dr2b = (const float*)d_in[22];
    const float* ct1w = (const float*)d_in[23]; const float* ct1b = (const float*)d_in[24];
    const float* ct2w = (const float*)d_in[25]; const float* ct2b = (const float*)d_in[26];
    const float* instw = (const float*)d_in[27]; const float* instb = (const float*)d_in[28];
    const float* adjl  = (const float*)d_in[29];
    const float* demb  = (const float*)d_in[30];
    const float* dg1w  = (const float*)d_in[31]; const float* dg1b = (const float*)d_in[32];
    const float* dg2w  = (const float*)d_in[33]; const float* dg2b = (const float*)d_in[34];

    // ---- workspace carve ----
    char* base = (char*)d_ws; size_t off = 0;
    auto alloc = [&](size_t bytes) -> void* {
        void* p = base + off; off = (off + bytes + 255) & ~(size_t)255; return p;
    };
    auto allocU = [&](long n) { return (unsigned short*)alloc((size_t)n * 2); };
    auto allocF = [&](long n) { return (float*)alloc((size_t)n * 4); };

    unsigned short* wb_ec1 = allocU(64 * 48);
    unsigned short* wb_ec2 = allocU(128 * 1024);
    unsigned short* wb_ec3 = allocU(128 * 1152);
    unsigned short* wb_er1 = allocU(2 * 32 * 1152);
    unsigned short* wb_er2 = allocU(2 * 128 * 32);
    unsigned short* wb_mu  = allocU((long)D * FLAT);
    unsigned short* wb_lv  = allocU((long)D * FLAT);
    unsigned short* wb_dfc = allocU((long)FLAT * D);
    unsigned short* wb_dr1 = allocU(2 * 32 * 1152);
    unsigned short* wb_dr2 = allocU(2 * 128 * 32);
    unsigned short* wb_ct1 = allocU(64 * 2048);
    unsigned short* wb_ct2 = allocU(3 * 1024);

    float* f_h1 = allocF((long)B * 64 * 32 * 32);   // also reused for dec_ct1 output
    float* f_h2 = allocF((long)B * 128 * 16 * 16);
    float* f_pa = allocF((long)B * FLAT);
    float* f_pb = allocF((long)B * FLAT);
    float* f_t  = allocF((long)B * 32 * 16 * 16);
    float* f_mu = allocF(B * D);   float* f_lv = allocF(B * D);
    float* f_xl = allocF(B * D);   float* f_v  = allocF(B * D);
    float* f_v2 = allocF(B * D);   float* f_z  = allocF(B * D);
    float* f_deg = allocF(256);    float* f_dinv = allocF(256);
    float* f_adj = allocF(D * D);  float* f_t0 = allocF(D * 64);
    float* f_xd  = allocF(D * 64); float* f_t2 = allocF(D);
    float* f_dimp = allocF(D);
    (void)ws_size; (void)n_in; (void)out_size;

    float* o_recon = (float*)d_out;
    const int RSZ = B * C * H * H;                 // 3145728
    float* o_elbo = o_recon + RSZ;
    float* o_nact = o_elbo + 1;
    float* o_dimp = o_nact + 1;
    float* o_vinst = o_dimp + D;

    auto gup = [](long n) { return (unsigned int)((n + 255) / 256); };

    // ---- weight prep (fp32 -> bf16 [Cout][Ktot]) ----
    wprep_copy<<<gup(64L * 48), 256, 0, stream>>>(ec1w, wb_ec1, 64L * 48);
    wprep_copy<<<gup(128L * 1024), 256, 0, stream>>>(ec2w, wb_ec2, 128L * 1024);
    wprep_copy<<<gup(128L * 1152), 256, 0, stream>>>(ec3w, wb_ec3, 128L * 1152);
    for (int i = 0; i < L; ++i) {
        wprep_copy<<<gup(32L * 1152), 256, 0, stream>>>(er1w + (long)i * 32 * 1152,
                                                        wb_er1 + (long)i * 32 * 1152, 32L * 1152);
        wprep_copy<<<gup(128L * 32), 256, 0, stream>>>(er2w + (long)i * 128 * 32,
                                                       wb_er2 + (long)i * 128 * 32, 128L * 32);
        wprep_copy<<<gup(32L * 1152), 256, 0, stream>>>(dr1w + (long)i * 32 * 1152,
                                                        wb_dr1 + (long)i * 32 * 1152, 32L * 1152);
        wprep_copy<<<gup(128L * 32), 256, 0, stream>>>(dr2w + (long)i * 128 * 32,
                                                       wb_dr2 + (long)i * 128 * 32, 128L * 32);
    }
    wprep_transpose<<<gup((long)FLAT * D), 256, 0, stream>>>(fmw, wb_mu, FLAT, D);
    wprep_transpose<<<gup((long)FLAT * D), 256, 0, stream>>>(flw, wb_lv, FLAT, D);
    wprep_transpose<<<gup((long)FLAT * D), 256, 0, stream>>>(dfw, wb_dfc, D, FLAT);
    wprep_hwio<<<gup(64L * 2048), 256, 0, stream>>>(ct1w, wb_ct1, 4, 4, 128, 64);
    wprep_hwio<<<gup(3L * 1024), 256, 0, stream>>>(ct2w, wb_ct2, 4, 4, 64, 3);

    // ---- encoder ----
    run_conv(stream, x,    wb_ec1, ec1b, nullptr, f_h1, 3, 64, 64, 64, 32, 32, 4, 4, 2, 1, 0, 0, 1, 0, B);
    run_conv(stream, f_h1, wb_ec2, ec2b, nullptr, f_h2, 64, 32, 32, 128, 16, 16, 4, 4, 2, 1, 0, 0, 1, 0, B);
    run_conv(stream, f_h2, wb_ec3, ec3b, nullptr, f_pa, 128, 16, 16, 128, 16, 16, 3, 3, 1, 1, 0, 0, 0, 0, B);
    float* pa = f_pa; float* pb = f_pb;
    for (int i = 0; i < L; ++i) {
        run_conv(stream, pa,  wb_er1 + (long)i * 32 * 1152, er1b + i * Rb, nullptr, f_t,
                 128, 16, 16, 32, 16, 16, 3, 3, 1, 1, 0, 1, 0, 0, B);
        run_conv(stream, f_t, wb_er2 + (long)i * 128 * 32, er2b + i * hid, pa, pb,
                 32, 16, 16, 128, 16, 16, 1, 1, 1, 0, 0, 1, 0, 1, B);
        float* tmp = pa; pa = pb; pb = tmp;
    }
    // FC mu / logvar (ReLU of res-stack tail fused into relu_in)
    run_conv(stream, pa, wb_mu, fmb, nullptr, f_mu, FLAT, 1, 1, D, 1, 1, 1, 1, 1, 0, 0, 1, 0, 0, B);
    run_conv(stream, pa, wb_lv, flb, nullptr, f_lv, FLAT, 1, 1, D, 1, 1, 1, 1, 1, 0, 0, 1, 0, 0, B);

    // ---- instance GNN (chain graph) ----
    deg_kernel<<<1, 256, 0, stream>>>(edst, E, f_deg, f_dinv, B);
    sgemm_naive<<<gup((long)B * D), 256, 0, stream>>>(f_mu, instw, nullptr, f_xl, B, D, D, 0);
    gcn_agg_kernel<<<gup((long)B * D), 256, 0, stream>>>(f_xl, esrc, edst, E, f_deg, f_dinv, instb, f_v, B, D);
    sgemm_naive<<<gup((long)B * D), 256, 0, stream>>>(f_v, instw + D * D, nullptr, f_xl, B, D, D, 0);
    gcn_agg_kernel<<<gup((long)B * D), 256, 0, stream>>>(f_xl, esrc, edst, E, f_deg, f_dinv, instb + D, f_v2, B, D);

    // ---- dim GNN ----
    dim_adj_kernel<<<1, 256, 0, stream>>>(adjl, f_adj, D);
    sgemm_naive<<<gup((long)D * 64), 256, 0, stream>>>(demb, dg1w, nullptr, f_t0, D, 64, 16, 0);
    sgemm_naive<<<gup((long)D * 64), 256, 0, stream>>>(f_adj, f_t0, dg1b, f_xd, D, 64, D, 1);
    sgemm_naive<<<gup((long)D), 256, 0, stream>>>(f_xd, dg2w, nullptr, f_t2, D, 1, 64, 0);
    sgemm_naive<<<gup((long)D), 256, 0, stream>>>(f_adj, f_t2, dg2b, f_dimp, D, 1, D, 0);

    // ---- reparameterize ----
    reparam_kernel<<<gup((long)B * D), 256, 0, stream>>>(f_mu, f_lv, eps, f_v2, f_dimp, f_z, o_vinst, B * D, D);
    copyf_kernel<<<1, 256, 0, stream>>>(f_dimp, o_dimp, D);

    // ---- decoder ----
    run_conv(stream, f_z, wb_dfc, dfb, nullptr, f_pa, D, 1, 1, FLAT, 1, 1, 1, 1, 1, 0, 0, 0, 0, 0, B);
    pa = f_pa; pb = f_pb;
    for (int i = 0; i < L; ++i) {
        run_conv(stream, pa,  wb_dr1 + (long)i * 32 * 1152, dr1b + i * Rb, nullptr, f_t,
                 128, 16, 16, 32, 16, 16, 3, 3, 1, 1, 0, 1, 0, 0, B);
        run_conv(stream, f_t, wb_dr2 + (long)i * 128 * 32, dr2b + i * hid, pa, pb,
                 32, 16, 16, 128, 16, 16, 1, 1, 1, 0, 0, 1, 0, 1, B);
        float* tmp = pa; pa = pb; pb = tmp;
    }
    // deconvs (x2 zero-upsample, virtual pad 2); res-stack tail ReLU fused into ct1 relu_in
    run_conv(stream, pa,   wb_ct1, ct1b, nullptr, f_h1,    128, 16, 16, 64, 32, 32, 4, 4, 1, 2, 1, 1, 1, 0, B);
    run_conv(stream, f_h1, wb_ct2, ct2b, nullptr, o_recon, 64, 32, 32, 3, 64, 64, 4, 4, 1, 2, 1, 0, 0, 0, B);

    // ---- ELBO + active dims (deterministic single-block reductions) ----
    elbo_kernel<<<1, 256, 0, stream>>>(o_recon, x, RSZ, f_z, f_lv, B * D, o_elbo);
    active_kernel<<<1, 256, 0, stream>>>(f_mu, f_lv, B, D, o_nact);
}
